// MambaLayer_12816182411895
// MI455X (gfx1250) — matrix operand order repaired and checked by reading, not compiled, not run on hardware
//
#include <hip/hip_runtime.h>

// ---------------------------------------------------------------------------
// MambaLayer for MI455X (gfx1250). All GEMMs via v_wmma_f32_16x16x32_bf16,
// with double-buffered async global->LDS tile copies (ASYNCcnt) when the
// toolchain exposes the gfx1250 async-LDS builtins.
// Sizes (from reference): B=4, L=512, DM=1024, DS=16, DC=4, DI=2048, DTR=64.
// ---------------------------------------------------------------------------

typedef __attribute__((ext_vector_type(16))) __bf16 v16bf;
typedef __attribute__((ext_vector_type(8)))  __bf16 v8bf;
typedef __attribute__((ext_vector_type(8)))  float  v8f;
typedef __attribute__((ext_vector_type(4)))  int    i32x4;

#define MROWS 2048   // B*L
#define DM_   1024
#define DI_   2048
#define DS_   16
#define DC_   4
#define DTR_  64
#define XPN   128    // x_proj output padded 96 -> 128
#define LSEQ  512
#define BATCH 4

#if defined(__has_builtin)
#if __has_builtin(__builtin_amdgcn_global_load_async_to_lds_b128) && \
    __has_builtin(__builtin_amdgcn_s_wait_asynccnt)
#define HAVE_ASYNC_LDS 1
#endif
#endif

#ifdef HAVE_ASYNC_LDS
#define GAS __attribute__((address_space(1)))
#define LAS __attribute__((address_space(3)))
// b128 async copy: builtin expects (i32x4 AS1*, i32x4 AS3*, imm offset, imm cpol)
#define ASYNC_B128(gptr, lptr) \
  __builtin_amdgcn_global_load_async_to_lds_b128((GAS i32x4*)(gptr), \
                                                 (LAS i32x4*)(lptr), 0, 0)
#endif

// ---------------------------------------------------------------------------
// fp32 -> bf16 conversion (for GEMM A operands)
// ---------------------------------------------------------------------------
__global__ __launch_bounds__(256) void k_f32_to_bf16(const float* __restrict__ in,
                                                     __bf16* __restrict__ out, int n) {
  int i = blockIdx.x * blockDim.x + threadIdx.x;
  if (i < n) out[i] = (__bf16)in[i];
}

// ---------------------------------------------------------------------------
// Weight prep: W [N][K] fp32 row-major  ->  Wt [K][Npad] bf16 row-major
// (zero-pads columns N..Npad-1)
// ---------------------------------------------------------------------------
__global__ __launch_bounds__(256) void k_transpose_bf16(const float* __restrict__ W,
                                                        __bf16* __restrict__ Wt,
                                                        int N, int K, int Npad) {
  int n = blockIdx.x * blockDim.x + threadIdx.x;
  int k = blockIdx.y;
  if (n < Npad) {
    float v = (n < N) ? W[(size_t)n * K + k] : 0.0f;
    Wt[(size_t)k * Npad + n] = (__bf16)v;
  }
}

// ---------------------------------------------------------------------------
// WMMA GEMM: C[M][N] = A[M][K](bf16) * Bw[K][N](bf16), f32 accumulate.
// Block tile 128x64, 8 waves, each wave 32x32 (2x2 WMMA 16x16x32 tiles).
// Async path: double-buffered LDS tiles filled by global_load_async_to_lds,
// overlap of tile(k+1) copy with tile(k) WMMA; s_wait_asynccnt + barrier.
// EPI: 0 = (+bias?) fp32 out; 1 = +bias, LeakyReLU(0.01), bf16 out;
//      2 = +bias, softplus, fp32 out.
// ---------------------------------------------------------------------------
template <int EPI>
__global__ __launch_bounds__(256) void k_gemm(const __bf16* __restrict__ A,
                                              const __bf16* __restrict__ Bw,
                                              const float* __restrict__ bias,
                                              float* __restrict__ Cf,
                                              __bf16* __restrict__ Cb,
                                              int Mdim, int Ndim, int Kdim) {
#ifdef HAVE_ASYNC_LDS
  __shared__ __bf16 sA[2][128][32];  // 16 KB (double buffered)
  __shared__ __bf16 sB[2][32][64];   //  8 KB
#else
  __shared__ __bf16 sA[1][128][32];  //  8 KB
  __shared__ __bf16 sB[1][32][64];   //  4 KB
#endif

  const int tid  = threadIdx.x;
  const int lane = tid & 31;
  const int wave = tid >> 5;
  const int bm = blockIdx.x * 128;
  const int bn = blockIdx.y * 64;
  const int wm = (wave >> 1) * 32;  // wave row offset within block tile
  const int wn = (wave & 1) * 32;   // wave col offset within block tile

  v8f acc[2][2] = {};

  // global staging addresses (per-thread)
  const int arow = tid >> 1;         // 0..127
  const int acol = (tid & 1) * 16;   // 0 or 16
  const int brow = tid >> 3;         // 0..31
  const int bcol = (tid & 7) * 8;    // 0..56
  const __bf16* gA = A  + (size_t)(bm + arow) * Kdim + acol;
  const __bf16* gB = Bw + (size_t)brow * Ndim + bn + bcol;

  // A fragment layout (ISA 7.12.2, 16-bit A 16x32): lane = (l&15) = M row,
  // elems 0..7 -> K = (l>>4)*8 + i, elems 8..15 -> K = 16 + (l>>4)*8 + i.
  const int ar    = lane & 15;
  const int abase = (lane >> 4) * 8;

#ifdef HAVE_ASYNC_LDS
  // -------- async double-buffered pipeline --------
  auto issue_tile = [&](int k0, int b) {
    const __bf16* ga = gA + k0;
    const __bf16* gb = gB + (size_t)k0 * Ndim;
    ASYNC_B128(ga,     &sA[b][arow][acol]);
    ASYNC_B128(ga + 8, &sA[b][arow][acol + 8]);
    ASYNC_B128(gb,     &sB[b][brow][bcol]);
  };

  issue_tile(0, 0);
  __builtin_amdgcn_s_wait_asynccnt(0);
  __syncthreads();

  int buf = 0;
  for (int k0 = 0; k0 < Kdim; k0 += 32) {
    if (k0 + 32 < Kdim) issue_tile(k0 + 32, buf ^ 1);  // overlap copy w/ WMMA
#pragma unroll
    for (int mi = 0; mi < 2; ++mi) {
      v8bf lo = *(const v8bf*)&sA[buf][wm + mi * 16 + ar][abase];
      v8bf hi = *(const v8bf*)&sA[buf][wm + mi * 16 + ar][16 + abase];
      v16bf a = __builtin_shufflevector(lo, hi, 0, 1, 2, 3, 4, 5, 6, 7,
                                        8, 9, 10, 11, 12, 13, 14, 15);
#pragma unroll
      for (int ni = 0; ni < 2; ++ni) {
        v16bf b;
#pragma unroll
        for (int s = 0; s < 16; ++s)
          b[s] = sB[buf][s < 8 ? abase + s : 16 + abase + (s - 8)][wn + ni * 16 + ar];
        acc[mi][ni] = __builtin_amdgcn_wmma_f32_16x16x32_bf16(
            false, a, false, b, (short)0, acc[mi][ni], false, false);
      }
    }
    __builtin_amdgcn_s_wait_asynccnt(0);
    __syncthreads();
    buf ^= 1;
  }
#else
  // -------- synchronous staged path (fallback) --------
  for (int k0 = 0; k0 < Kdim; k0 += 32) {
    *(v16bf*)&sA[0][arow][acol] = *(const v16bf*)(gA + k0);
    *(v8bf*)&sB[0][brow][bcol]  = *(const v8bf*)(gB + (size_t)k0 * Ndim);
    if (k0 + 32 < Kdim) {
      __builtin_prefetch(gA + k0 + 32, 0, 0);
      __builtin_prefetch(gB + (size_t)(k0 + 32) * Ndim, 0, 0);
    }
    __syncthreads();
#pragma unroll
    for (int mi = 0; mi < 2; ++mi) {
      v8bf lo = *(const v8bf*)&sA[0][wm + mi * 16 + ar][abase];
      v8bf hi = *(const v8bf*)&sA[0][wm + mi * 16 + ar][16 + abase];
      v16bf a = __builtin_shufflevector(lo, hi, 0, 1, 2, 3, 4, 5, 6, 7,
                                        8, 9, 10, 11, 12, 13, 14, 15);
#pragma unroll
      for (int ni = 0; ni < 2; ++ni) {
        v16bf b;
#pragma unroll
        for (int s = 0; s < 16; ++s)
          b[s] = sB[0][s < 8 ? abase + s : 16 + abase + (s - 8)][wn + ni * 16 + ar];
        acc[mi][ni] = __builtin_amdgcn_wmma_f32_16x16x32_bf16(
            false, a, false, b, (short)0, acc[mi][ni], false, false);
      }
    }
    __syncthreads();
  }
#endif

  // C/D layout (ISA 7.12.2): lanes 0-15 N=lane M=r; lanes 16-31 N=lane-16 M=r+8.
  const int rhalf = (lane >> 4) * 8;
  const int ccol  = lane & 15;
#pragma unroll
  for (int mi = 0; mi < 2; ++mi) {
#pragma unroll
    for (int ni = 0; ni < 2; ++ni) {
      const int col = bn + wn + ni * 16 + ccol;
      const float bv = bias ? bias[col] : 0.0f;
#pragma unroll
      for (int r = 0; r < 8; ++r) {
        const int row = bm + wm + mi * 16 + rhalf + r;
        float v = acc[mi][ni][r] + bv;
        if (EPI == 0) {
          Cf[(size_t)row * Ndim + col] = v;
        } else if (EPI == 1) {       // LeakyReLU -> bf16
          v = (v >= 0.0f) ? v : 0.01f * v;
          Cb[(size_t)row * Ndim + col] = (__bf16)v;
        } else {                     // softplus -> fp32
          v = (v > 20.0f) ? v : log1pf(__expf(v));
          Cf[(size_t)row * Ndim + col] = v;
        }
      }
    }
  }
}

// ---------------------------------------------------------------------------
// Depthwise causal conv (DC=4) + bias + SiLU over xs half of xz.
// xz: [M][2*DI]; writes xs fp32 [M][DI] and xs bf16 [M][DI].
// ---------------------------------------------------------------------------
__global__ __launch_bounds__(256) void k_conv_silu(const float* __restrict__ xz,
                                                   const float* __restrict__ cw,
                                                   const float* __restrict__ cb,
                                                   float* __restrict__ xs,
                                                   __bf16* __restrict__ xsbf) {
  int idx = blockIdx.x * blockDim.x + threadIdx.x;  // over M*DI
  int d = idx & (DI_ - 1);
  int m = idx >> 11;
  int l = m & (LSEQ - 1);
  int b = m >> 9;
  float acc = cb[d];
#pragma unroll
  for (int j = 0; j < DC_; ++j) {
    int ls = l - (DC_ - 1) + j;
    if (ls >= 0)
      acc += xz[((size_t)(b * LSEQ + ls)) * (2 * DI_) + d] * cw[d * DC_ + j];
  }
  float s = acc / (1.0f + __expf(-acc));  // SiLU
  xs[idx] = s;
  xsbf[idx] = (__bf16)s;
}

// ---------------------------------------------------------------------------
// Extract dt_r (cols 0..63 of padded x_dbl [M][128]) as bf16 [M][64].
// ---------------------------------------------------------------------------
__global__ __launch_bounds__(256) void k_extract_dtr(const float* __restrict__ xdbl,
                                                     __bf16* __restrict__ dtr) {
  int i = blockIdx.x * blockDim.x + threadIdx.x;  // over M*DTR
  int m = i >> 6;
  int c = i & (DTR_ - 1);
  dtr[i] = (__bf16)xdbl[(size_t)m * XPN + c];
}

// ---------------------------------------------------------------------------
// Selective scan: one thread per (batch, channel d). h[16] in registers,
// 512 serial steps; B/C read as float4 (broadcast across the block -> WGP$).
// ---------------------------------------------------------------------------
__global__ __launch_bounds__(256) void k_scan(const float* __restrict__ dt,
                                              const float* __restrict__ xs,
                                              const float* __restrict__ xdbl,
                                              const float* __restrict__ A_log,
                                              float* __restrict__ y) {
  const int b = blockIdx.x >> 3;                        // DI/256 = 8 blocks/batch
  const int d = ((blockIdx.x & 7) << 8) + threadIdx.x;  // channel
  float A[DS_], h[DS_];
#pragma unroll
  for (int n = 0; n < DS_; ++n) {
    A[n] = -__expf(A_log[d * DS_ + n]);
    h[n] = 0.0f;
  }
  for (int l = 0; l < LSEQ; ++l) {
    const int m = b * LSEQ + l;
    const float4* bc = (const float4*)(xdbl + (size_t)m * XPN + DTR_);
    float Bv[DS_], Cv[DS_];
    ((float4*)Bv)[0] = bc[0]; ((float4*)Bv)[1] = bc[1];
    ((float4*)Bv)[2] = bc[2]; ((float4*)Bv)[3] = bc[3];
    ((float4*)Cv)[0] = bc[4]; ((float4*)Cv)[1] = bc[5];
    ((float4*)Cv)[2] = bc[6]; ((float4*)Cv)[3] = bc[7];
    const float dtv = dt[(size_t)m * DI_ + d];
    const float xv  = xs[(size_t)m * DI_ + d];
    const float dx  = dtv * xv;
    float acc = 0.0f;
#pragma unroll
    for (int n = 0; n < DS_; ++n) {
      h[n] = __expf(dtv * A[n]) * h[n] + dx * Bv[n];
      acc += h[n] * Cv[n];
    }
    y[(size_t)m * DI_ + d] = acc;
  }
}

// ---------------------------------------------------------------------------
// Gate: yg = (y + D*xs) * silu(z), z = xz[:, DI:2*DI]. bf16 out for out_proj.
// ---------------------------------------------------------------------------
__global__ __launch_bounds__(256) void k_gate(const float* __restrict__ y,
                                              const float* __restrict__ xs,
                                              const float* __restrict__ xz,
                                              const float* __restrict__ Dv,
                                              __bf16* __restrict__ yg) {
  int i = blockIdx.x * blockDim.x + threadIdx.x;  // over M*DI
  int d = i & (DI_ - 1);
  int m = i >> 11;
  float z = xz[(size_t)m * (2 * DI_) + DI_ + d];
  float g = z / (1.0f + __expf(-z));
  yg[i] = (__bf16)((y[i] + Dv[d] * xs[i]) * g);
}

// ---------------------------------------------------------------------------
// Residual add + LayerNorm over DM=1024 cols; one block per row.
// Writes fp32 (and optionally bf16 for the next GEMM's A operand).
// ---------------------------------------------------------------------------
__global__ __launch_bounds__(256) void k_add_ln(const float* __restrict__ a,
                                                const float* __restrict__ res,
                                                const float* __restrict__ w,
                                                const float* __restrict__ bias,
                                                float* __restrict__ outF,
                                                __bf16* __restrict__ outB) {
  const int m = blockIdx.x;
  __shared__ float red[2][8];
  const float* pa = a + (size_t)m * DM_;
  const float* pr = res + (size_t)m * DM_;
  float v[4], s = 0.0f, s2 = 0.0f;
#pragma unroll
  for (int j = 0; j < 4; ++j) {
    int c = threadIdx.x + j * 256;
    float t = pa[c] + pr[c];
    v[j] = t; s += t; s2 += t * t;
  }
#pragma unroll
  for (int off = 16; off; off >>= 1) {  // wave32 reduction
    s  += __shfl_xor(s, off, 32);
    s2 += __shfl_xor(s2, off, 32);
  }
  if ((threadIdx.x & 31) == 0) {
    red[0][threadIdx.x >> 5] = s;
    red[1][threadIdx.x >> 5] = s2;
  }
  __syncthreads();
  float ts = 0.0f, ts2 = 0.0f;
#pragma unroll
  for (int i = 0; i < 8; ++i) { ts += red[0][i]; ts2 += red[1][i]; }
  const float mean = ts * (1.0f / DM_);
  const float var  = ts2 * (1.0f / DM_) - mean * mean;
  const float rstd = rsqrtf(var + 1e-12f);
#pragma unroll
  for (int j = 0; j < 4; ++j) {
    int c = threadIdx.x + j * 256;
    float o = (v[j] - mean) * rstd * w[c] + bias[c];
    outF[(size_t)m * DM_ + c] = o;
    if (outB) outB[(size_t)m * DM_ + c] = (__bf16)o;
  }
}

// ---------------------------------------------------------------------------
// Host-side orchestration
// ---------------------------------------------------------------------------
extern "C" void kernel_launch(void* const* d_in, const int* in_sizes, int n_in,
                              void* d_out, int out_size, void* d_ws, size_t ws_size,
                              hipStream_t stream) {
  const float* x         = (const float*)d_in[0];
  const float* in_proj_w = (const float*)d_in[1];
  const float* conv_w    = (const float*)d_in[2];
  const float* conv_b    = (const float*)d_in[3];
  const float* x_proj_w  = (const float*)d_in[4];
  const float* dt_proj_w = (const float*)d_in[5];
  const float* dt_proj_b = (const float*)d_in[6];
  const float* A_log     = (const float*)d_in[7];
  const float* Dvec      = (const float*)d_in[8];
  const float* out_proj_w= (const float*)d_in[9];
  const float* ln1_w     = (const float*)d_in[10];
  const float* ln1_b     = (const float*)d_in[11];
  const float* ffn_w1    = (const float*)d_in[12];
  const float* ffn_b1    = (const float*)d_in[13];
  const float* ffn_w2    = (const float*)d_in[14];
  const float* ffn_b2    = (const float*)d_in[15];
  const float* ln2_w     = (const float*)d_in[16];
  const float* ln2_b     = (const float*)d_in[17];

  char* ws = (char*)d_ws;
  size_t off = 0;
  auto alloc = [&](size_t bytes) -> char* {
    char* p = ws + off;
    off = (off + bytes + 255) & ~(size_t)255;
    return p;
  };

  __bf16* xbf   = (__bf16*)alloc((size_t)MROWS * DM_ * 2);
  __bf16* Wi_t  = (__bf16*)alloc((size_t)DM_ * (2 * DI_) * 2);   // [1024][4096]
  __bf16* Wxp_t = (__bf16*)alloc((size_t)DI_ * XPN * 2);         // [2048][128]
  __bf16* Wdt_t = (__bf16*)alloc((size_t)DTR_ * DI_ * 2);        // [64][2048]
  __bf16* Wo_t  = (__bf16*)alloc((size_t)DI_ * DM_ * 2);         // [2048][1024]
  __bf16* W1_t  = (__bf16*)alloc((size_t)DM_ * (2 * DM_) * 2);   // [1024][2048]
  __bf16* W2_t  = (__bf16*)alloc((size_t)(2 * DM_) * DM_ * 2);   // [2048][1024]
  float*  xz    = (float*) alloc((size_t)MROWS * (2 * DI_) * 4); // 32 MB
  float*  xs    = (float*) alloc((size_t)MROWS * DI_ * 4);
  __bf16* xsbf  = (__bf16*)alloc((size_t)MROWS * DI_ * 2);
  float*  xdbl  = (float*) alloc((size_t)MROWS * XPN * 4);
  __bf16* dtrbf = (__bf16*)alloc((size_t)MROWS * DTR_ * 2);
  float*  dt    = (float*) alloc((size_t)MROWS * DI_ * 4);
  float*  yb    = (float*) alloc((size_t)MROWS * DI_ * 4);
  __bf16* ygbf  = (__bf16*)alloc((size_t)MROWS * DI_ * 2);
  float*  mo    = (float*) alloc((size_t)MROWS * DM_ * 4);
  float*  h1f   = (float*) alloc((size_t)MROWS * DM_ * 4);
  __bf16* h1bf  = (__bf16*)alloc((size_t)MROWS * DM_ * 2);
  __bf16* fbf   = (__bf16*)alloc((size_t)MROWS * (2 * DM_) * 2);
  float*  ff    = (float*) alloc((size_t)MROWS * DM_ * 4);

  // --- prep: x -> bf16; weights -> bf16 [K][Npad] ---
  k_f32_to_bf16<<<(MROWS * DM_) / 256, 256, 0, stream>>>(x, xbf, MROWS * DM_);
  k_transpose_bf16<<<dim3(16, DM_), 256, 0, stream>>>(in_proj_w, Wi_t, 2 * DI_, DM_, 2 * DI_);
  k_transpose_bf16<<<dim3(1, DI_), 256, 0, stream>>>(x_proj_w, Wxp_t, DTR_ + 2 * DS_, DI_, XPN);
  k_transpose_bf16<<<dim3(8, DTR_), 256, 0, stream>>>(dt_proj_w, Wdt_t, DI_, DTR_, DI_);
  k_transpose_bf16<<<dim3(4, DI_), 256, 0, stream>>>(out_proj_w, Wo_t, DM_, DI_, DM_);
  k_transpose_bf16<<<dim3(8, DM_), 256, 0, stream>>>(ffn_w1, W1_t, 2 * DM_, DM_, 2 * DM_);
  k_transpose_bf16<<<dim3(4, 2 * DM_), 256, 0, stream>>>(ffn_w2, W2_t, DM_, 2 * DM_, DM_);

  // --- in_proj: xz = x @ Wi^T   (2048 x 4096 x 1024) ---
  k_gemm<0><<<dim3(16, 64), 256, 0, stream>>>(xbf, Wi_t, nullptr, xz, nullptr,
                                              MROWS, 2 * DI_, DM_);
  // --- depthwise conv + SiLU ---
  k_conv_silu<<<(MROWS * DI_) / 256, 256, 0, stream>>>(xz, conv_w, conv_b, xs, xsbf);
  // --- x_proj: x_dbl = xs @ Wxp^T (padded N=128) ---
  k_gemm<0><<<dim3(16, 2), 256, 0, stream>>>(xsbf, Wxp_t, nullptr, xdbl, nullptr,
                                             MROWS, XPN, DI_);
  // --- dt_r -> bf16; dt = softplus(dt_r @ Wdt^T + b) ---
  k_extract_dtr<<<(MROWS * DTR_) / 256, 256, 0, stream>>>(xdbl, dtrbf);
  k_gemm<2><<<dim3(16, 32), 256, 0, stream>>>(dtrbf, Wdt_t, dt_proj_b, dt, nullptr,
                                              MROWS, DI_, DTR_);
  // --- selective scan (serial over L, parallel over B*DI) ---
  k_scan<<<BATCH * (DI_ / 256), 256, 0, stream>>>(dt, xs, xdbl, A_log, yb);
  // --- gating ---
  k_gate<<<(MROWS * DI_) / 256, 256, 0, stream>>>(yb, xs, xz, Dvec, ygbf);
  // --- out_proj ---
  k_gemm<0><<<dim3(16, 16), 256, 0, stream>>>(ygbf, Wo_t, nullptr, mo, nullptr,
                                              MROWS, DM_, DI_);
  // --- residual + LN1 ---
  k_add_ln<<<MROWS, 256, 0, stream>>>(mo, x, ln1_w, ln1_b, h1f, h1bf);
  // --- FFN1: bias + LeakyReLU -> bf16 ---
  k_gemm<1><<<dim3(16, 32), 256, 0, stream>>>(h1bf, W1_t, ffn_b1, nullptr, fbf,
                                              MROWS, 2 * DM_, DM_);
  // --- FFN2: + bias -> fp32 ---
  k_gemm<0><<<dim3(16, 16), 256, 0, stream>>>(fbf, W2_t, ffn_b2, ff, nullptr,
                                              MROWS, DM_, 2 * DM_);
  // --- residual + LN2 -> d_out ---
  k_add_ln<<<MROWS, 256, 0, stream>>>(ff, h1f, ln2_w, ln2_b, (float*)d_out, nullptr);
}
